// SSD_47339129536581
// MI455X (gfx1250) — matrix operand order repaired and checked by reading, not compiled, not run on hardware
//
#include <hip/hip_runtime.h>
#include <hip/hip_bf16.h>
#include <math.h>

// ---------------------------------------------------------------------------
// SSD300 forward for MI455X (gfx1250).  All 35 convolutions are implicit-GEMM
// on V_WMMA_F32_16X16X32_BF16 (wave32):
//   - activations NHWC bf16 (conv1 input channel-padded to 32)
//   - weights pre-packed in exact WMMA A-fragment lane order (bf16)
//   - A fragment: one 32B contiguous load per cout sub-tile (L2-resident)
//   - B fragment: ONE 32B contiguous load per pixel group (NHWC!)
//   - 8 WMMAs per K-chunk (4 cout sub-tiles x 2 pixel sub-tiles)
//   - epilogue: packed 16B bf16 stores (8 consecutive couts per VGPR group)
// No LDS, no barriers in the conv kernel.
// ---------------------------------------------------------------------------

typedef __attribute__((ext_vector_type(16))) __bf16          v16bf;
typedef __attribute__((ext_vector_type(8)))  float           v8f;
typedef __attribute__((ext_vector_type(8)))  unsigned short  us8;

__device__ __forceinline__ unsigned short f2bf(float f) {
  unsigned int u = __float_as_uint(f);
  u += 0x7FFFu + ((u >> 16) & 1u);          // round-to-nearest-even
  return (unsigned short)(u >> 16);
}
__device__ __forceinline__ float bf2f(unsigned short s) {
  return __uint_as_float(((unsigned int)s) << 16);
}

// ---------------- input: NCHW fp32 -> NHWC bf16, channels padded ----------
__global__ void ssd_x_to_nhwc(const float* __restrict__ in,
                              unsigned short* __restrict__ out,
                              int N, int C, int Cp, int HW) {
  int idx = blockIdx.x * blockDim.x + threadIdx.x;
  int total = N * HW * Cp;
  if (idx >= total) return;
  int c   = idx % Cp;
  int pix = (idx / Cp) % HW;
  int n   = idx / (Cp * HW);
  float v = (c < C) ? in[((size_t)n * C + c) * HW + pix] : 0.f;
  out[idx] = f2bf(v);
}

// ---------------- weight repack: OIHW fp32 -> WMMA A-fragment bf16 ---------
// Packed layout: [tile16 T][chunk c][lane 0..31][elem 0..15]
//   chunk c = (ky*KW + kx)*nCc + c0/32,  nCc = ceil(Cin/32)
//   lane: M = lane%16; K-local = kb + e, kb = (lane>>4)*16
__global__ void ssd_pack_w(const float* __restrict__ w,
                           unsigned short* __restrict__ pk,
                           int Cout, int Cin, int KH, int KW,
                           int nChunks, int nCc, int total) {
  int idx = blockIdx.x * blockDim.x + threadIdx.x;
  if (idx >= total) return;
  int e    = idx & 15;
  int lane = (idx >> 4) & 31;
  int rest = idx >> 9;
  int c    = rest % nChunks;
  int T    = rest / nChunks;
  int c0   = (c % nCc) * 32;
  int kk   = c / nCc;
  int ky = kk / KW, kx = kk - ky * KW;
  int kb = (lane >> 4) * 16;
  int klocal = kb + e;
  int co = T * 16 + (lane & 15);
  int ci = c0 + klocal;
  float v = 0.f;
  if (co < Cout && ci < Cin)
    v = w[((size_t)(co * Cin + ci) * KH + ky) * KW + kx];
  pk[idx] = f2bf(v);
}

// ---------------- generic max-pool (NHWC bf16) -----------------------------
__global__ void ssd_maxpool(const unsigned short* __restrict__ in,
                            unsigned short* __restrict__ out,
                            int N, int C, int H, int W,
                            int k, int s, int padlo, int Ho, int Wo) {
  int idx = blockIdx.x * blockDim.x + threadIdx.x;
  int total = N * C * Ho * Wo;
  if (idx >= total) return;
  int c = idx % C; int t = idx / C;
  int ox = t % Wo;  t /= Wo;
  int oy = t % Ho;  int n = t / Ho;
  int y0 = oy * s - padlo, x0 = ox * s - padlo;
  float m = -INFINITY;
  for (int dy = 0; dy < k; ++dy) {
    int iy = y0 + dy; if (iy < 0 || iy >= H) continue;
    for (int dx = 0; dx < k; ++dx) {
      int ix = x0 + dx; if (ix < 0 || ix >= W) continue;
      m = fmaxf(m, bf2f(in[((size_t)(n * H + iy) * W + ix) * C + c]));
    }
  }
  out[idx] = f2bf(m);
}

// ---------------- L2Norm over channels (wave-per-pixel, NHWC) --------------
__global__ __launch_bounds__(256)
void ssd_l2norm(const unsigned short* __restrict__ in,
                const float* __restrict__ w,
                unsigned short* __restrict__ out,
                int N, int C, int HW) {
  int wv   = (blockIdx.x * blockDim.x + threadIdx.x) >> 5;
  int lane = threadIdx.x & 31;
  int total = N * HW * HW;
  if (wv >= total) return;
  size_t base = (size_t)wv * C;                // wv == n*HW*HW + pix
  float ss = 0.f;
  for (int c = lane; c < C; c += 32) {
    float v = bf2f(in[base + c]); ss += v * v;
  }
  for (int off = 16; off > 0; off >>= 1) ss += __shfl_xor(ss, off, 32);
  float inv = 1.f / (sqrtf(ss) + 1e-10f);
  for (int c = lane; c < C; c += 32)
    out[base + c] = f2bf(bf2f(in[base + c]) * inv * w[c]);
}

// ---------------- implicit-GEMM conv via WMMA (NHWC, no LDS) ---------------
// Wave tile: 64 couts x 32 pixels (8 acc).  Block = 8 independent waves.
// mode 0: bf16 NHWC activation (+bias, ReLU); requires Cout % 64 == 0
// mode 1: head -> fp32 scatter into d_out NHWC-flatten region (guarded).
__global__ __launch_bounds__(256)
void ssd_conv_wmma(const unsigned short* __restrict__ in,
                   const unsigned short* __restrict__ wpk,
                   const float* __restrict__ bias,
                   unsigned short* __restrict__ outAct,
                   float* __restrict__ outHead,
                   int N, int Cin, int H, int W,
                   int Cout, int KH, int KW,
                   int stride, int pad, int dil,
                   int Ho, int Wo, int nChunks,
                   int relu, int mode, int perN, int srcOff, int Cs) {
  const int tid  = threadIdx.x;
  const int wv   = tid >> 5;
  const int lane = tid & 31;
  const int laneHalf = lane >> 4;
  const int lmod = lane & 15;

  const int n       = blockIdx.z;
  const int cbase   = blockIdx.y * 64;                // 4 tiles of 16 couts
  const int Npix    = Ho * Wo;
  const int pixBase = blockIdx.x * 256 + wv * 32;     // 2 pixel sub-tiles
  const size_t inImg = (size_t)H * W * Cin;

  // per-lane pixel decomposition (one divide per sub-tile, hoisted)
  int yb[2], xb[2]; bool pv[2];
#pragma unroll
  for (int g = 0; g < 2; ++g) {
    int p = pixBase + g * 16 + lmod;
    if (p < Npix) {
      int oy = p / Wo, ox = p - oy * Wo;
      yb[g] = oy * stride - pad; xb[g] = ox * stride - pad; pv[g] = true;
    } else { yb[g] = 0; xb[g] = 0; pv[g] = false; }
  }

  const int kbB = laneHalf * 16;                      // B K-range base
  const unsigned short* ibase = in + (size_t)n * inImg;
  const unsigned short* wbase =
      wpk + ((size_t)(cbase >> 4) * nChunks) * 512 + lane * 16;
  const size_t tstride = (size_t)nChunks * 512;       // per cout sub-tile

  v8f acc[8] = {};                                    // [g*4 + t]

  int chunk = 0;
  for (int ky = 0; ky < KH; ++ky) {
    for (int kx = 0; kx < KW; ++kx) {
      const int kyd = ky * dil, kxd = kx * dil;
      int  o[2]; bool ok[2];
#pragma unroll
      for (int g = 0; g < 2; ++g) {
        int iy = yb[g] + kyd, ix = xb[g] + kxd;
        ok[g] = pv[g] && (unsigned)iy < (unsigned)H && (unsigned)ix < (unsigned)W;
        o[g]  = (iy * W + ix) * Cin;                  // NHWC pixel offset
      }
      for (int c0 = 0; c0 < Cin; c0 += 32, ++chunk) {
        // ---- B fragments: ONE 32B contiguous NHWC load per pixel group
        v16bf bf0 = {}, bf1 = {};
        const int cc = c0 + kbB;
        if (ok[0]) bf0 = *(const v16bf*)(ibase + (size_t)o[0] + cc);
        if (ok[1]) bf1 = *(const v16bf*)(ibase + (size_t)o[1] + cc);
        // ---- A fragments: one 32B load per sub-tile; 8 WMMAs
        const unsigned short* ap = wbase + (size_t)chunk * 512;
        __builtin_prefetch(ap + 512, 0, 1);           // next K-chunk
#pragma unroll
        for (int t = 0; t < 4; ++t) {
          v16bf a = *(const v16bf*)(ap + tstride * t);
          acc[t]     = __builtin_amdgcn_wmma_f32_16x16x32_bf16(
              false, a, false, bf0, (short)0, acc[t],     false, false);
          acc[4 + t] = __builtin_amdgcn_wmma_f32_16x16x32_bf16(
              false, a, false, bf1, (short)0, acc[4 + t], false, false);
        }
      }
    }
  }

  // ---- epilogue: D VGPR r -> cout = cbase + t*16 + 8*laneHalf + r (contig!)
#pragma unroll
  for (int g = 0; g < 2; ++g) {
    int pix = pixBase + g * 16 + lmod;
    if (pix >= Npix) continue;
    if (mode == 0) {                                  // Cout % 64 == 0
      unsigned short* op = outAct + ((size_t)n * Npix + pix) * Cout;
#pragma unroll
      for (int t = 0; t < 4; ++t) {
        int co0 = cbase + t * 16 + 8 * laneHalf;
        us8 pkv;
#pragma unroll
        for (int r = 0; r < 8; ++r) {
          float v = acc[g * 4 + t][r] + bias[co0 + r];
          if (relu) v = fmaxf(v, 0.f);
          pkv[r] = f2bf(v);
        }
        *(us8*)(op + co0) = pkv;                      // 16B packed store
      }
    } else {
      float* hp = outHead + (size_t)n * perN + srcOff + (size_t)pix * Cs;
#pragma unroll
      for (int t = 0; t < 4; ++t)
#pragma unroll
        for (int r = 0; r < 8; ++r) {
          int co = cbase + t * 16 + 8 * laneHalf + r;
          if (co < Cout) hp[co] = acc[g * 4 + t][r] + bias[co];
        }
    }
  }
}

// ---------------------------------------------------------------------------
// Host orchestration
// ---------------------------------------------------------------------------
static inline int conv_out_dim(int H, int K, int s, int p, int d) {
  return (H + 2 * p - d * (K - 1) - 1) / s + 1;
}
static inline int pack_tiles(int Cout) { return ((Cout + 63) / 64) * 4; }

struct PkConv { unsigned short* pk; int nChunks; };

static PkConv pack_weights(hipStream_t st, const float* w,
                           int Cout, int Cin, int K,
                           unsigned char* base, size_t& off) {
  int nCc = (Cin + 31) / 32;
  int nChunks = K * K * nCc;
  int total = pack_tiles(Cout) * nChunks * 512;
  unsigned short* pk = (unsigned short*)(base + off);
  off += (size_t)total * 2;
  ssd_pack_w<<<(total + 255) / 256, 256, 0, st>>>(w, pk, Cout, Cin, K, K,
                                                  nChunks, nCc, total);
  PkConv r; r.pk = pk; r.nChunks = nChunks; return r;
}

static void launch_conv(hipStream_t st, const unsigned short* in,
                        const PkConv& pw, const float* b,
                        int N, int Cin, int H, int Cout,
                        int K, int s, int p, int d,
                        unsigned short* out, int relu) {
  int Ho = conv_out_dim(H, K, s, p, d);
  dim3 grid((Ho * Ho + 255) / 256, (Cout + 63) / 64, N);
  ssd_conv_wmma<<<grid, 256, 0, st>>>(in, pw.pk, b, out, nullptr,
                                      N, Cin, H, H, Cout, K, K, s, p, d,
                                      Ho, Ho, pw.nChunks, relu, 0, 0, 0, 0);
}

static void launch_head(hipStream_t st, const unsigned short* in,
                        const PkConv& pw, const float* b,
                        int N, int Cin, int HW, int Cout,
                        float* outBase, int perN, int srcOff) {
  dim3 grid((HW * HW + 255) / 256, (Cout + 63) / 64, N);
  ssd_conv_wmma<<<grid, 256, 0, st>>>(in, pw.pk, b, nullptr, outBase,
                                      N, Cin, HW, HW, Cout, 3, 3, 1, 1, 1,
                                      HW, HW, pw.nChunks, 0, 1, perN, srcOff, Cout);
}

static void launch_pool(hipStream_t st, const unsigned short* in,
                        unsigned short* out, int N, int C, int H,
                        int k, int s, int padlo, int Ho) {
  int total = N * C * Ho * Ho;
  ssd_maxpool<<<(total + 255) / 256, 256, 0, st>>>(in, out, N, C, H, H,
                                                   k, s, padlo, Ho, Ho);
}

extern "C" void kernel_launch(void* const* d_in, const int* in_sizes, int n_in,
                              void* d_out, int out_size, void* d_ws, size_t ws_size,
                              hipStream_t stream) {
  (void)in_sizes; (void)n_in; (void)out_size;
  // Input order: x, vgg_w[15], vgg_b[15], l2_weight, extra_w[8], extra_b[8],
  //              loc_w[6], loc_b[6], conf_w[6], conf_b[6], dbox  (73 entries)
  const float* x = (const float*)d_in[0];
  const float* vw[15]; const float* vb[15];
  for (int i = 0; i < 15; ++i) { vw[i] = (const float*)d_in[1 + i];
                                 vb[i] = (const float*)d_in[16 + i]; }
  const float* l2w = (const float*)d_in[31];
  const float* ew[8]; const float* eb[8];
  for (int i = 0; i < 8; ++i) { ew[i] = (const float*)d_in[32 + i];
                                eb[i] = (const float*)d_in[40 + i]; }
  const float* lw[6]; const float* lb[6]; const float* cw[6]; const float* cb[6];
  for (int i = 0; i < 6; ++i) { lw[i] = (const float*)d_in[48 + i];
                                lb[i] = (const float*)d_in[54 + i];
                                cw[i] = (const float*)d_in[60 + i];
                                cb[i] = (const float*)d_in[66 + i]; }
  const float* dbox = (const float*)d_in[72];
  float* out = (float*)d_out;

  // ---- workspace layout: bf16 NHWC ping-pong + 6 sources + packed weights
  const size_t ACT = 46080000;                 // 4*300*300*64 * 2B (max act)
  const size_t OFF_A  = 0;
  const size_t OFF_B  = ACT;
  const size_t OFF_S1 = 2 * ACT;
  const size_t OFF_S2 = OFF_S1 + 5914624ull;   // 4*38*38*512*2
  const size_t OFF_S3 = OFF_S2 + 2957312ull;   // 4*19*19*1024*2
  const size_t OFF_S4 = OFF_S3 + 409600ull;    // 4*10*10*512*2
  const size_t OFF_S5 = OFF_S4 + 51200ull;     // 4*5*5*256*2
  const size_t OFF_S6 = OFF_S5 + 18432ull;     // 4*3*3*256*2
  const size_t OFF_PK = OFF_S6 + 2048ull;      // packed weights start here
  const size_t PK_BYTES = 27817984ull * 2ull;  // 35 layers packed (bf16)
  if (ws_size < OFF_PK + PK_BYTES) return;

  unsigned char* ws = (unsigned char*)d_ws;
  unsigned short* bufA = (unsigned short*)(ws + OFF_A);
  unsigned short* bufB = (unsigned short*)(ws + OFF_B);
  unsigned short* s1   = (unsigned short*)(ws + OFF_S1);
  unsigned short* s2   = (unsigned short*)(ws + OFF_S2);
  unsigned short* s3   = (unsigned short*)(ws + OFF_S3);
  unsigned short* s4   = (unsigned short*)(ws + OFF_S4);
  unsigned short* s5   = (unsigned short*)(ws + OFF_S5);
  unsigned short* s6   = (unsigned short*)(ws + OFF_S6);

  // ---- pre-pack all conv weights into WMMA A-fragment layout (bf16)
  size_t pko = OFF_PK;
  const int vC[15][2] = {{64,3},{64,64},{128,64},{128,128},{256,128},{256,256},
                         {256,256},{512,256},{512,512},{512,512},{512,512},
                         {512,512},{512,512},{1024,512},{1024,1024}};
  const int vK[15] = {3,3,3,3,3,3,3,3,3,3,3,3,3,3,1};
  PkConv pvgg[15];
  for (int i = 0; i < 15; ++i)
    pvgg[i] = pack_weights(stream, vw[i], vC[i][0], vC[i][1], vK[i], ws, pko);
  const int eC[8][2] = {{256,1024},{512,256},{128,512},{256,128},
                        {128,256},{256,128},{128,256},{256,128}};
  const int eK[8] = {1,3,1,3,1,3,1,3};
  PkConv pext[8];
  for (int i = 0; i < 8; ++i)
    pext[i] = pack_weights(stream, ew[i], eC[i][0], eC[i][1], eK[i], ws, pko);
  const int SC[6] = {512, 1024, 512, 256, 256, 256};
  const int NB[6] = {4, 6, 6, 6, 4, 4};
  PkConv ploc[6], pconf[6];
  for (int i = 0; i < 6; ++i) {
    ploc[i]  = pack_weights(stream, lw[i], NB[i] * 4,  SC[i], 3, ws, pko);
    pconf[i] = pack_weights(stream, cw[i], NB[i] * 21, SC[i], 3, ws, pko);
  }

  // ---- input fp32 NCHW -> bf16 NHWC, channels padded 3 -> 32
  {
    int total = 4 * 300 * 300 * 32;
    ssd_x_to_nhwc<<<(total + 255) / 256, 256, 0, stream>>>(x, bufA, 4, 3, 32, 90000);
  }

  // ---- VGG backbone (conv1 uses padded Cin=32)
  launch_conv(stream, bufA, pvgg[0],  vb[0],  4,  32, 300,  64, 3, 1, 1, 1, bufB, 1);
  launch_conv(stream, bufB, pvgg[1],  vb[1],  4,  64, 300,  64, 3, 1, 1, 1, bufA, 1);
  launch_pool(stream, bufA, bufB, 4,  64, 300, 2, 2, 0, 150);
  launch_conv(stream, bufB, pvgg[2],  vb[2],  4,  64, 150, 128, 3, 1, 1, 1, bufA, 1);
  launch_conv(stream, bufA, pvgg[3],  vb[3],  4, 128, 150, 128, 3, 1, 1, 1, bufB, 1);
  launch_pool(stream, bufB, bufA, 4, 128, 150, 2, 2, 0, 75);
  launch_conv(stream, bufA, pvgg[4],  vb[4],  4, 128,  75, 256, 3, 1, 1, 1, bufB, 1);
  launch_conv(stream, bufB, pvgg[5],  vb[5],  4, 256,  75, 256, 3, 1, 1, 1, bufA, 1);
  launch_conv(stream, bufA, pvgg[6],  vb[6],  4, 256,  75, 256, 3, 1, 1, 1, bufB, 1);
  launch_pool(stream, bufB, bufA, 4, 256,  75, 2, 2, 0, 38);   // ceil-mode 75->38
  launch_conv(stream, bufA, pvgg[7],  vb[7],  4, 256,  38, 512, 3, 1, 1, 1, bufB, 1);
  launch_conv(stream, bufB, pvgg[8],  vb[8],  4, 512,  38, 512, 3, 1, 1, 1, bufA, 1);
  launch_conv(stream, bufA, pvgg[9],  vb[9],  4, 512,  38, 512, 3, 1, 1, 1, bufB, 1);

  // L2Norm of conv4_3 -> source 1
  {
    int waves = 4 * 38 * 38;
    ssd_l2norm<<<(waves * 32 + 255) / 256, 256, 0, stream>>>(bufB, l2w, s1, 4, 512, 38);
  }

  launch_pool(stream, bufB, bufA, 4, 512,  38, 2, 2, 0, 19);
  launch_conv(stream, bufA, pvgg[10], vb[10], 4, 512,  19, 512, 3, 1, 1, 1, bufB, 1);
  launch_conv(stream, bufB, pvgg[11], vb[11], 4, 512,  19, 512, 3, 1, 1, 1, bufA, 1);
  launch_conv(stream, bufA, pvgg[12], vb[12], 4, 512,  19, 512, 3, 1, 1, 1, bufB, 1);
  launch_pool(stream, bufB, bufA, 4, 512,  19, 3, 1, 1, 19);   // pool5 k3 s1 p1
  launch_conv(stream, bufA, pvgg[13], vb[13], 4, 512,  19, 1024, 3, 1, 6, 6, bufB, 1); // conv6
  launch_conv(stream, bufB, pvgg[14], vb[14], 4, 1024, 19, 1024, 1, 1, 0, 1, s2,   1); // conv7

  // ---- extras
  launch_conv(stream, s2,   pext[0], eb[0], 4, 1024, 19, 256, 1, 1, 0, 1, bufA, 1);
  launch_conv(stream, bufA, pext[1], eb[1], 4,  256, 19, 512, 3, 2, 1, 1, s3,   1);
  launch_conv(stream, s3,   pext[2], eb[2], 4,  512, 10, 128, 1, 1, 0, 1, bufA, 1);
  launch_conv(stream, bufA, pext[3], eb[3], 4,  128, 10, 256, 3, 2, 1, 1, s4,   1);
  launch_conv(stream, s4,   pext[4], eb[4], 4,  256,  5, 128, 1, 1, 0, 1, bufA, 1);
  launch_conv(stream, bufA, pext[5], eb[5], 4,  128,  5, 256, 3, 1, 0, 1, s5,   1);
  launch_conv(stream, s5,   pext[6], eb[6], 4,  256,  3, 128, 1, 1, 0, 1, bufA, 1);
  launch_conv(stream, bufA, pext[7], eb[7], 4,  128,  3, 256, 3, 1, 0, 1, s6,   1);

  // ---- detection heads (fp32 scatter directly into d_out)
  // d_out: loc [4*8732*4=139712] | conf [4*8732*21=733488] | dbox [34928]
  const int HWs[6] = {38, 19, 10, 5, 3, 1};
  const unsigned short* srcs[6] = {s1, s2, s3, s4, s5, s6};
  const int locOff[6]  = {0, 23104, 31768, 34168, 34768, 34912};
  const int confOff[6] = {0, 121296, 166782, 179382, 182532, 183288};
  float* locBase  = out;
  float* confBase = out + 139712;
  for (int s = 0; s < 6; ++s) {
    launch_head(stream, srcs[s], ploc[s],  lb[s], 4, SC[s], HWs[s], NB[s] * 4,
                locBase, 34928, locOff[s]);
    launch_head(stream, srcs[s], pconf[s], cb[s], 4, SC[s], HWs[s], NB[s] * 21,
                confBase, 183372, confOff[s]);
  }

  // ---- dbox passthrough
  hipMemcpyAsync(out + 873200, dbox, 34928 * sizeof(float),
                 hipMemcpyDeviceToDevice, stream);
}
